// HyperAggLayer_52467320487985
// MI455X (gfx1250) — compile-verified, matrix operand order checked by reading, not run on hardware
//
#include <hip/hip_runtime.h>
#include <hip/hip_bf16.h>

typedef __attribute__((ext_vector_type(16))) __bf16 v16bf;
typedef __attribute__((ext_vector_type(8)))  float  v8f;

#define DD 128
#define HH 4
#define HD 512

// ---- monotone float<->uint encoding for atomicMax on signed floats ----
__device__ __forceinline__ unsigned enc_f32(float f) {
  unsigned u = __float_as_uint(f);
  return (u & 0x80000000u) ? ~u : (u | 0x80000000u);
}
__device__ __forceinline__ float dec_f32(unsigned u) {
  return (u & 0x80000000u) ? __uint_as_float(u & 0x7FFFFFFFu) : __uint_as_float(~u);
}

// ---- kernel 0: init output, segment sums, encoded segment maxima ----
__global__ void k_init(float* out, int outN, float* segSum, unsigned* segMax, int nh) {
  int i = blockIdx.x * blockDim.x + threadIdx.x;
  int stride = gridDim.x * blockDim.x;
  for (int t = i; t < outN; t += stride) out[t] = 0.f;
  for (int t = i; t < nh; t += stride) { segSum[t] = 0.f; segMax[t] = 0u; }
}

// ---- kernel 1: fold attention vectors into W -> P_src/P_dst [128][4] ----
__global__ void k_proj(const float* __restrict__ Wsrc, const float* __restrict__ Wdst,
                       const float* __restrict__ attS, const float* __restrict__ attD,
                       float* __restrict__ P) {
  int t = threadIdx.x;           // 512 threads: k = t&127, h = t>>7
  int k = t & 127, h = t >> 7;
  float as = 0.f, ad = 0.f;
  for (int d = 0; d < DD; ++d) {
    as += Wsrc[k * HD + h * DD + d] * attS[h * DD + d];
    ad += Wdst[k * HD + h * DD + d] * attD[h * DD + d];
  }
  P[k * HH + h] = as;
  P[HD + k * HH + h] = ad;
}

// ---- kernel 2: per-edge attention logits + leaky relu + segment max ----
__global__ void k_logits(const float* __restrict__ src, const float* __restrict__ tgt,
                         const long long* __restrict__ ei, const float* __restrict__ P,
                         float* __restrict__ alpha, unsigned* __restrict__ segMax, int E) {
  __shared__ float sPs[HD], sPd[HD];
  for (int t = threadIdx.x; t < HD; t += blockDim.x) { sPs[t] = P[t]; sPd[t] = P[HD + t]; }
  __syncthreads();
  int stride = gridDim.x * blockDim.x;
  for (int e = blockIdx.x * blockDim.x + threadIdx.x; e < E; e += stride) {
    const float4* sr = (const float4*)(src + (size_t)e * DD);
    const float4* tr = (const float4*)(tgt + (size_t)e * DD);
    float acc[HH] = {0.f, 0.f, 0.f, 0.f};
#pragma unroll 4
    for (int kq = 0; kq < DD / 4; ++kq) {
      float4 a = sr[kq], b = tr[kq];
      int k = kq * 4;
#pragma unroll
      for (int h = 0; h < HH; ++h) {
        acc[h] += a.x * sPs[(k + 0) * HH + h] + a.y * sPs[(k + 1) * HH + h]
                + a.z * sPs[(k + 2) * HH + h] + a.w * sPs[(k + 3) * HH + h]
                + b.x * sPd[(k + 0) * HH + h] + b.y * sPd[(k + 1) * HH + h]
                + b.z * sPd[(k + 2) * HH + h] + b.w * sPd[(k + 3) * HH + h];
      }
    }
    long long s = ei[(size_t)E + e];  // edge_index[1][e]
#pragma unroll
    for (int h = 0; h < HH; ++h) {
      float v = acc[h];
      v = v > 0.f ? v : 0.2f * v;      // leaky relu 0.2
      alpha[(size_t)e * HH + h] = v;
      atomicMax(&segMax[(size_t)s * HH + h], enc_f32(v));
    }
  }
}

// ---- kernel 3: exp(logit - segmax), accumulate segment denominators ----
__global__ void k_expw(const long long* __restrict__ ei, float* __restrict__ alpha,
                       const unsigned* __restrict__ segMax, float* __restrict__ segSum, int E) {
  int stride = gridDim.x * blockDim.x;
  int total = E * HH;
  for (int t = blockIdx.x * blockDim.x + threadIdx.x; t < total; t += stride) {
    int e = t >> 2, h = t & 3;
    long long s = ei[(size_t)E + e];
    float m = dec_f32(segMax[(size_t)s * HH + h]);
    float w = __expf(alpha[t] - m);
    alpha[t] = w;
    atomicAdd(&segSum[(size_t)s * HH + h], w);
  }
}

// Fragment-layout swizzle: K-local (0..31) -> (half, elem)
__device__ __forceinline__ void frag_idx(int kl, int& hf, int& el) {
  hf = (kl >> 3) & 1;
  el = (kl & 7) + ((kl >> 4) & 1) * 8;
}

// ---- kernel 4: bf16 WMMA GEMM (x_i = src @ W_src) fused with alpha-scale
// ---- and scatter-add into v_out. Full W (128KB bf16) in LDS, fragment
// layout. 32-edge M tile: each B fragment feeds 2 WMMAs (two M-subtiles),
// 8 waves x 64 cols, 8 independent accumulator chains per wave.
__global__ void __launch_bounds__(256)
k_gemm_scatter(const float* __restrict__ src, const long long* __restrict__ ei,
               const float* __restrict__ W, const float* __restrict__ alpha,
               const float* __restrict__ segSum, float* __restrict__ out, int E) {
  // sW: [kk(4)][half(2)][c(512)][elem(16)] = 65536 bf16 = 128 KB
  // sA: [kk(4)][half(2)][m(32)][elem(16)]  =  4096 bf16 =   8 KB
  __shared__ alignas(32) __bf16 sW[4 * 2 * 512 * 16];
  __shared__ alignas(32) __bf16 sA[4 * 2 * 32 * 16];
  __shared__ alignas(16) int    sSeg[32];
  __shared__ alignas(16) float  sScaleT[HH * 32];   // [h][m]

  // stage full W, swizzled to fragment layout (amortized over ~10 tiles)
#pragma unroll 2
  for (int i = 0; i < 256; ++i) {
    int lin = i * 256 + threadIdx.x;   // 0..65535
    int k = lin >> 9, c = lin & 511;
    int kk = k >> 5, hf, el;
    frag_idx(k & 31, hf, el);
    sW[((((kk * 2 + hf) * 512) + c) << 4) + el] = (__bf16)W[k * HD + c];
  }

  const int lane = threadIdx.x & 31;
  const int wave = threadIdx.x >> 5;
  const int half = lane >> 4;
  const int l16  = lane & 15;
  const int c0   = wave * 64;          // this wave's 4 column tiles

  const v16bf* sWf = (const v16bf*)sW; // index: (kk*2+half)*512 + c
  const v16bf* sAf = (const v16bf*)sA; // index: ((kk*2+half)*32 + m)

  const int numTiles = (E + 31) >> 5;
  for (int tile = blockIdx.x; tile < numTiles; tile += gridDim.x) {
    __syncthreads();                   // protect sA/sSeg/sScaleT (also first W use)
    const int e0 = tile << 5;
    // stage A tile (32 edges x 128 K) as bf16, fragment-swizzled
#pragma unroll
    for (int i = 0; i < 16; ++i) {
      int lin = i * 256 + threadIdx.x; // 0..4095
      int r = lin >> 7, k = lin & 127;
      int e = e0 + r;
      float v = (e < E) ? src[(size_t)e * DD + k] : 0.f;
      int kk = k >> 5, hf, el;
      frag_idx(k & 31, hf, el);
      sA[((((kk * 2 + hf) * 32) + r) << 4) + el] = (__bf16)v;
    }
    // stage seg ids and normalized alpha scales, transposed [h][m]
    if (threadIdx.x < 128) {
      int r = threadIdx.x >> 2, h = threadIdx.x & 3;
      int e = e0 + r;
      if (e < E) {
        long long s = ei[(size_t)E + e];
        if (h == 0) sSeg[r] = (int)s;
        float w = alpha[(size_t)e * HH + h];
        float den = segSum[(size_t)s * HH + h];
        sScaleT[h * 32 + r] = w / (den + 1e-16f) * 0.25f;
      } else {
        if (h == 0) sSeg[r] = 0;
        sScaleT[h * 32 + r] = 0.f;
      }
    }
    __syncthreads();

    v8f acc0[4] = {v8f{}, v8f{}, v8f{}, v8f{}};  // M-subtile 0 (edges 0..15)
    v8f acc1[4] = {v8f{}, v8f{}, v8f{}, v8f{}};  // M-subtile 1 (edges 16..31)
#pragma unroll
    for (int kk = 0; kk < 4; ++kk) {   // K = 128 in 4 steps of 32
      const int kh = kk * 2 + half;
      v16bf af0 = sAf[kh * 32 + l16];
      v16bf af1 = sAf[kh * 32 + 16 + l16];
#pragma unroll
      for (int nt = 0; nt < 4; ++nt) {
        v16bf b = sWf[kh * 512 + c0 + nt * 16 + l16];
        acc0[nt] = __builtin_amdgcn_wmma_f32_16x16x32_bf16(false, af0, false, b,
                                                           (short)0, acc0[nt], false, false);
        acc1[nt] = __builtin_amdgcn_wmma_f32_16x16x32_bf16(false, af1, false, b,
                                                           (short)0, acc1[nt], false, false);
      }
    }

    // scatter: C/D layout -> lane col n=l16, VGPR r -> row m = r + 8*half
    if (e0 + 32 <= E) {                // hot path: full tile, no per-elem guards
#pragma unroll
      for (int ms = 0; ms < 2; ++ms) {
        const int mBase = ms * 16 + half * 8;
        int4 sgA = *(const int4*)&sSeg[mBase];
        int4 sgB = *(const int4*)&sSeg[mBase + 4];
#pragma unroll
        for (int nt = 0; nt < 4; ++nt) {
          int Cabs = c0 + nt * 16 + l16;
          int h = Cabs >> 7;           // wave-uniform per nt (tiles head-aligned)
          int d = Cabs & 127;
          float4 scA = *(const float4*)&sScaleT[h * 32 + mBase];
          float4 scB = *(const float4*)&sScaleT[h * 32 + mBase + 4];
          v8f a = ms ? acc1[nt] : acc0[nt];
          atomicAdd(out + (unsigned)sgA.x * (unsigned)DD + d, a[0] * scA.x);
          atomicAdd(out + (unsigned)sgA.y * (unsigned)DD + d, a[1] * scA.y);
          atomicAdd(out + (unsigned)sgA.z * (unsigned)DD + d, a[2] * scA.z);
          atomicAdd(out + (unsigned)sgA.w * (unsigned)DD + d, a[3] * scA.w);
          atomicAdd(out + (unsigned)sgB.x * (unsigned)DD + d, a[4] * scB.x);
          atomicAdd(out + (unsigned)sgB.y * (unsigned)DD + d, a[5] * scB.y);
          atomicAdd(out + (unsigned)sgB.z * (unsigned)DD + d, a[6] * scB.z);
          atomicAdd(out + (unsigned)sgB.w * (unsigned)DD + d, a[7] * scB.w);
        }
      }
    } else {                           // cold tail tile: guarded
#pragma unroll
      for (int ms = 0; ms < 2; ++ms) {
        const int mBase = ms * 16 + half * 8;
#pragma unroll
        for (int nt = 0; nt < 4; ++nt) {
          int Cabs = c0 + nt * 16 + l16;
          int h = Cabs >> 7;
          int d = Cabs & 127;
          v8f a = ms ? acc1[nt] : acc0[nt];
#pragma unroll
          for (int r = 0; r < 8; ++r) {
            int m = r + mBase;
            if (e0 + m < E) {
              float v = a[r] * sScaleT[h * 32 + m];
              atomicAdd(out + (unsigned)sSeg[m] * (unsigned)DD + d, v);
            }
          }
        }
      }
    }
  }
}

extern "C" void kernel_launch(void* const* d_in, const int* in_sizes, int n_in,
                              void* d_out, int out_size, void* d_ws, size_t ws_size,
                              hipStream_t stream) {
  const float*     src  = (const float*)d_in[0];
  const float*     tgt  = (const float*)d_in[1];
  const long long* ei   = (const long long*)d_in[2];
  const float*     Wsrc = (const float*)d_in[3];
  const float*     Wdst = (const float*)d_in[4];
  const float*     attS = (const float*)d_in[5];
  const float*     attD = (const float*)d_in[6];
  float* out = (float*)d_out;

  const int E = in_sizes[0] / DD;
  const int N = out_size / DD;

  // workspace layout: alpha[E*H] | segMax[N*H] | segSum[N*H] | P[1024]
  char* wsb = (char*)d_ws;
  float*    wsAlpha = (float*)wsb;
  unsigned* wsMax   = (unsigned*)(wsb + (size_t)E * HH * 4);
  float*    wsSum   = (float*)(wsb + (size_t)E * HH * 4 + (size_t)N * HH * 4);
  float*    wsP     = (float*)(wsb + (size_t)E * HH * 4 + 2 * (size_t)N * HH * 4);

  k_init  <<<1024, 256, 0, stream>>>(out, N * DD, wsSum, wsMax, N * HH);
  k_proj  <<<1,    512, 0, stream>>>(Wsrc, Wdst, attS, attD, wsP);
  k_logits<<<1024, 256, 0, stream>>>(src, tgt, ei, wsP, wsAlpha, wsMax, E);
  k_expw  <<<1024, 256, 0, stream>>>(ei, wsAlpha, wsMax, wsSum, E);
  k_gemm_scatter<<<640, 256, 0, stream>>>(src, ei, Wsrc, wsAlpha, wsSum, out, E);
}
